// Encoder_17549236371982
// MI455X (gfx1250) — matrix operand order repaired and checked by reading, not compiled, not run on hardware
//
#include <hip/hip_runtime.h>
#include <math.h>

// ---------------- problem constants ----------------
#define B_    2
#define NV    6
#define FD    256
#define FH    28
#define FW    60
#define KK    (FH*FW)      // 1680 image tokens per view
#define DIM   128
#define HEADS 4
#define DH    32
#define QQ    2500         // 50*50 BEV queries
#define QT    157          // ceil(2500/16)
#define SCALE_F 0.17677669529663689f   // 32^-0.5

typedef _Float16 f16_t;
typedef _Float16 __attribute__((ext_vector_type(16))) v16h;
typedef float    __attribute__((ext_vector_type(8)))  v8f;

// ---------------- small helpers ----------------
__device__ __forceinline__ float blk_sum(float v, float* red) {
  int t = threadIdx.x;
  red[t] = v; __syncthreads();
#pragma unroll
  for (int s = 64; s > 0; s >>= 1) {
    if (t < s) red[t] += red[t + s];
    __syncthreads();
  }
  float r = red[0];
  __syncthreads();
  return r;
}

// f32 -> f16 conversion (for weight matrices)
__global__ void cvt_f16_kernel(const float* __restrict__ s, f16_t* __restrict__ d, int n) {
  int i = blockIdx.x * 256 + threadIdx.x;
  if (i < n) d[i] = (f16_t)s[i];
}

// c_embed[bn, d] = W_cam[d,:] . E_inv[bn, :, 3]
__global__ void cembed_kernel(const float* __restrict__ W_cam, const float* __restrict__ E_inv,
                              float* __restrict__ c_embed) {
  int bn = blockIdx.x, d = threadIdx.x;
  float s = 0.f;
#pragma unroll
  for (int i = 0; i < 4; ++i) s += W_cam[d*4 + i] * E_inv[bn*16 + i*4 + 3];
  c_embed[bn*DIM + d] = s;
}

// eval-BN + ReLU for both param sets, transposed to [ (b n p), c ] f16 rows for WMMA GEMM
__global__ void frelu_kernel(const float* __restrict__ feat,
                             const float* __restrict__ pg, const float* __restrict__ pb,
                             const float* __restrict__ pm, const float* __restrict__ pv,
                             const float* __restrict__ lg, const float* __restrict__ lb,
                             const float* __restrict__ lm, const float* __restrict__ lv,
                             f16_t* __restrict__ fp, f16_t* __restrict__ fl, int total) {
  int idx = blockIdx.x * 256 + threadIdx.x;
  if (idx >= total) return;
  int p  = idx % KK;
  int c  = (idx / KK) % FD;
  int bn = idx / (KK * FD);
  float f  = feat[idx];
  float s1 = pg[c] * rsqrtf(pv[c] + 1e-5f);
  float y1 = fmaxf((f - pm[c]) * s1 + pb[c], 0.f);
  float s2 = lg[c] * rsqrtf(lv[c] + 1e-5f);
  float y2 = fmaxf((f - lm[c]) * s2 + lb[c], 0.f);
  size_t o = ((size_t)bn * KK + p) * FD + c;
  fp[o] = (f16_t)y1;
  fl[o] = (f16_t)y2;
}

// query = normalize_d(W_bev@grid + b_bev - c_embed) + x ; then q_ln -> f16 row
__global__ __launch_bounds__(128)
void query_kernel(const float* __restrict__ x, const float* __restrict__ bev_grid,
                  const float* __restrict__ W_bev, const float* __restrict__ b_bev,
                  const float* __restrict__ c_embed,
                  const float* __restrict__ g, const float* __restrict__ bta,
                  f16_t* __restrict__ qln) {
  __shared__ float red[128];
  int bnq = blockIdx.x;
  int q  = bnq % QQ;
  int bn = bnq / QQ;
  int b  = bn / NV;
  int d  = threadIdx.x;
  float gx = bev_grid[q], gy = bev_grid[QQ + q];
  float wv  = W_bev[d*2] * gx + W_bev[d*2 + 1] * gy + b_bev[d];
  float bev = wv - c_embed[bn*DIM + d];
  float ss  = blk_sum(bev * bev, red);
  float qv  = bev / (sqrtf(ss) + 1e-7f) + x[((size_t)b*DIM + d)*QQ + q];
  float mean = blk_sum(qv, red) * (1.0f / DIM);
  float ctr  = qv - mean;
  float var  = blk_sum(ctr * ctr, red) * (1.0f / DIM);
  qln[((size_t)bn*QQ + q)*DIM + d] = (f16_t)(ctr * rsqrtf(var + 1e-5f) * g[d] + bta[d]);
}

// key = normalize_d(W_img@(E_inv@[I_inv@pix,1]) - c_embed) + keyfeat ; k_ln -> f16
__global__ __launch_bounds__(128)
void key_ln_kernel(const float* __restrict__ I_inv, const float* __restrict__ E_inv,
                   const float* __restrict__ image_plane, const float* __restrict__ W_img,
                   const float* __restrict__ c_embed, const float* __restrict__ keyfeat,
                   const float* __restrict__ g, const float* __restrict__ bta,
                   f16_t* __restrict__ kln) {
  __shared__ float red[128];
  int bnp = blockIdx.x;
  int p  = bnp % KK;
  int bn = bnp / KK;
  int d  = threadIdx.x;
  float ip0 = image_plane[p], ip1 = image_plane[KK + p], ip2 = image_plane[2*KK + p];
  const float* Ii = I_inv + bn*9;
  float c0 = Ii[0]*ip0 + Ii[1]*ip1 + Ii[2]*ip2;
  float c1 = Ii[3]*ip0 + Ii[4]*ip1 + Ii[5]*ip2;
  float c2 = Ii[6]*ip0 + Ii[7]*ip1 + Ii[8]*ip2;
  const float* Ei = E_inv + bn*16;
  float dv0 = Ei[0]*c0  + Ei[1]*c1  + Ei[2]*c2  + Ei[3];
  float dv1 = Ei[4]*c0  + Ei[5]*c1  + Ei[6]*c2  + Ei[7];
  float dv2 = Ei[8]*c0  + Ei[9]*c1  + Ei[10]*c2 + Ei[11];
  float dv3 = Ei[12]*c0 + Ei[13]*c1 + Ei[14]*c2 + Ei[15];
  float img = W_img[d*4]*dv0 + W_img[d*4+1]*dv1 + W_img[d*4+2]*dv2 + W_img[d*4+3]*dv3
            - c_embed[bn*DIM + d];
  float ss = blk_sum(img * img, red);
  float kv = img / (sqrtf(ss) + 1e-7f) + keyfeat[(size_t)bnp*DIM + d];
  float mean = blk_sum(kv, red) * (1.0f / DIM);
  float ctr  = kv - mean;
  float var  = blk_sum(ctr * ctr, red) * (1.0f / DIM);
  kln[(size_t)bnp*DIM + d] = (f16_t)(ctr * rsqrtf(var + 1e-5f) * g[d] + bta[d]);
}

// generic per-row LayerNorm: f32 [rows, DIM] -> f16
__global__ __launch_bounds__(128)
void row_ln_kernel(const float* __restrict__ src, const float* __restrict__ g,
                   const float* __restrict__ bta, f16_t* __restrict__ dst) {
  __shared__ float red[128];
  size_t row = blockIdx.x;
  int d = threadIdx.x;
  float v = src[row*DIM + d];
  float mean = blk_sum(v, red) * (1.0f / DIM);
  float ctr  = v - mean;
  float var  = blk_sum(ctr * ctr, red) * (1.0f / DIM);
  dst[row*DIM + d] = (f16_t)(ctr * rsqrtf(var + 1e-5f) * g[d] + bta[d]);
}

// z2 = pre_ln(z1 + skip(x)) ; emit f32 (residual) and f16 (MLP GEMM input)
__global__ __launch_bounds__(128)
void addskip_preln_kernel(const float* __restrict__ z1, const float* __restrict__ x,
                          const float* __restrict__ g, const float* __restrict__ bta,
                          float* __restrict__ z2, f16_t* __restrict__ z2h) {
  __shared__ float red[128];
  int row = blockIdx.x;
  int q = row % QQ, b = row / QQ, d = threadIdx.x;
  float v = z1[(size_t)row*DIM + d] + x[((size_t)b*DIM + d)*QQ + q];
  float mean = blk_sum(v, red) * (1.0f / DIM);
  float ctr  = v - mean;
  float var  = blk_sum(ctr * ctr, red) * (1.0f / DIM);
  float o = ctr * rsqrtf(var + 1e-5f) * g[d] + bta[d];
  z2[(size_t)row*DIM + d]  = o;
  z2h[(size_t)row*DIM + d] = (f16_t)o;
}

// out = post_ln(z2 + z3), transposed back to (b, DIM, H, W)
__global__ __launch_bounds__(128)
void final_kernel(const float* __restrict__ z2, const float* __restrict__ z3,
                  const float* __restrict__ g, const float* __restrict__ bta,
                  float* __restrict__ out) {
  __shared__ float red[128];
  int row = blockIdx.x;
  int q = row % QQ, b = row / QQ, d = threadIdx.x;
  float v = z2[(size_t)row*DIM + d] + z3[(size_t)row*DIM + d];
  float mean = blk_sum(v, red) * (1.0f / DIM);
  float ctr  = v - mean;
  float var  = blk_sum(ctr * ctr, red) * (1.0f / DIM);
  out[((size_t)b*DIM + d)*QQ + q] = ctr * rsqrtf(var + 1e-5f) * g[d] + bta[d];
}

// ---------------- WMMA NT-GEMM: C[M,N] = A[M,Kc] * W[N,Kc]^T (+bias, +gelu) ----------------
// A, W row-major f16; one wave computes a 16x64 strip (4 WMMA tiles, A reused 4x); fp32 acc.
// N must be a multiple of 64 (all call sites use 128 or 256).
__global__ __launch_bounds__(256)
void gemm_nt_kernel(const f16_t* __restrict__ A, int lda,
                    const f16_t* __restrict__ Bw, int ldb,
                    const float* __restrict__ bias,
                    float* __restrict__ Cf, f16_t* __restrict__ Ch,
                    int M, int N, int Kc, int act) {
  int wave = threadIdx.x >> 5;
  int lane = threadIdx.x & 31;
  int hgrp = lane >> 4;       // which 16-lane half
  int lrow = lane & 15;
  int n4 = N >> 6;            // 64-wide strips per row
  long tileId = (long)blockIdx.x * 8 + wave;
  long totalTiles = (long)((M + 15) >> 4) * n4;
  if (tileId >= totalTiles) return;           // wave-uniform: EXEC stays full for WMMA
  int mt  = (int)(tileId / n4);
  int nt4 = (int)(tileId % n4);
  int row = mt*16 + lrow; if (row >= M) row = M - 1;   // clamp, mask at store
  const f16_t* ap = A + (size_t)row * lda;
  const f16_t* bp[4];
#pragma unroll
  for (int j = 0; j < 4; ++j)
    bp[j] = Bw + (size_t)(nt4*64 + j*16 + lrow) * ldb;

  v8f acc[4] = {};
  for (int k0 = 0; k0 < Kc; k0 += 32) {
    __builtin_prefetch(ap + k0 + 32, 0, 1);      // global_prefetch_b8 next A chunk
    __builtin_prefetch(bp[0] + k0 + 32, 0, 1);   // next B chunk (cacheline pull)
    v16h a, bm[4];
#pragma unroll
    for (int e = 0; e < 16; ++e) {
      int ka = e + (hgrp << 4);                          // A layout K split
      a[e] = ap[k0 + ka];
      int kb = k0 + e + (hgrp << 4);                     // B layout K split
      bm[0][e] = bp[0][kb];
      bm[1][e] = bp[1][kb];
      bm[2][e] = bp[2][kb];
      bm[3][e] = bp[3][kb];
    }
#pragma unroll
    for (int j = 0; j < 4; ++j)
      acc[j] = __builtin_amdgcn_wmma_f32_16x16x32_f16(false, a, false, bm[j],
                                                      (short)0, acc[j], false, false);
  }
#pragma unroll
  for (int j = 0; j < 4; ++j) {
    int cc = nt4*64 + j*16 + lrow;
#pragma unroll
    for (int r = 0; r < 8; ++r) {
      int rr = mt*16 + r + hgrp*8;
      if (rr < M) {
        float v = acc[j][r] + (bias ? bias[cc] : 0.0f);
        if (act == 1) v = 0.5f * v * (1.0f + erff(v * 0.70710678118f)); // exact GELU
        if (Cf) Cf[(size_t)rr * N + cc] = v;
        if (Ch) Ch[(size_t)rr * N + cc] = (f16_t)v;
      }
    }
  }
}

// ---------------- fused flash attention ----------------
// qh: (B, NV, QQ, 128) f16 (inner = head*32 + d); kh/vh: (B, NV, KK, 128) f16.
// Softmax over the flattened (view, key) axis = 10080 keys; query differs per view.
// One wave per (b, head, 16-query tile). DH=32 == WMMA K for f16: 1 WMMA per 16x16 S tile.
// V chunks are DMA'd into LDS with GLOBAL_LOAD_ASYNC_TO_LDS_B128 (ASYNCcnt path).
#define PBUF_ELEMS (16*32)
#define VBUF_ELEMS (32*32)
#define VBUF_BYTE_OFF (PBUF_ELEMS*2)   // byte offset of vbuf inside the (sole) LDS block

__global__ __launch_bounds__(32)
void attn_kernel(const f16_t* __restrict__ qh, const f16_t* __restrict__ kh,
                 const f16_t* __restrict__ vh, f16_t* __restrict__ aout) {
  // Single shared allocation => assumed to start at wave LDS offset 0 (used for the
  // async-to-LDS destination addresses below).
  __shared__ __align__(16) f16_t smem[PBUF_ELEMS + VBUF_ELEMS];
  f16_t* pbuf = smem;                // 16x32 P repack staging
  f16_t* vbuf = smem + PBUF_ELEMS;   // 32(keys) x 32(dh) V chunk staging

  int tile = blockIdx.x;
  int qt = tile % QT; tile /= QT;
  int m  = tile % HEADS;
  int b  = tile / HEADS;
  int lane = threadIdx.x;
  int hgrp = lane >> 4;
  int lrow = lane & 15;

  v8f o0 = {}, o1 = {};                 // O tile: 16 q-rows x 32 dh (two 16x16 C tiles)
  float rmax[8], rsum[8];
#pragma unroll
  for (int r = 0; r < 8; ++r) { rmax[r] = -3.0e38f; rsum[r] = 0.f; }

  for (int n = 0; n < NV; ++n) {
    // Q tile A-operand for this view (query embedding depends on the camera)
    v16h aq;
    {
      int q = qt*16 + lrow; if (q >= QQ) q = QQ - 1;
      const f16_t* qp = qh + ((size_t)(b*NV + n)*QQ + q)*DIM + m*DH;
#pragma unroll
      for (int e = 0; e < 16; ++e) {
        int k = e + (hgrp << 4);
        aq[e] = qp[k];
      }
    }
    const f16_t* kbase = kh + (size_t)(b*NV + n)*KK*DIM + m*DH;
    const f16_t* vbase = vh + (size_t)(b*NV + n)*KK*DIM + m*DH;

    for (int kc = 0; kc < KK; kc += 32) {
      // ---- async-stage the V chunk (32 keys x 32 dh f16 = 2 KB) into LDS:
      //      4 x b128, one 16-byte slice per lane per issue; tracked by ASYNCcnt.
#pragma unroll
      for (int i = 0; i < 4; ++i) {
        int c   = i*32 + lane;           // 16-byte slice id, 0..127
        int key = c >> 2;                // 0..31 relative key row
        int dh0 = (c & 3) * 8;           // f16 start within the 32-wide dh row
        int kg  = kc + key; if (kg >= KK) kg = KK - 1;   // clamped; p==0 for masked keys
        const f16_t* gp = vbase + (size_t)kg * DIM + dh0;
        unsigned ldsa = VBUF_BYTE_OFF + (unsigned)c * 16u;
        asm volatile("global_load_async_to_lds_b128 %0, %1, off"
                     :: "v"(ldsa), "v"(gp) : "memory");
      }

      // ---- S = (Q * K^T) * scale : two 16x16 tiles (keys kc..+15, kc+16..+31)
      __builtin_prefetch(kbase + (size_t)(kc + 32 + lrow) * DIM, 0, 1); // next K chunk
      v16h bk0, bk1;
      {
        int k0c = kc + lrow;       if (k0c >= KK) k0c = KK - 1;
        int k1c = kc + 16 + lrow;  if (k1c >= KK) k1c = KK - 1;
        const f16_t* kp0 = kbase + (size_t)k0c * DIM;
        const f16_t* kp1 = kbase + (size_t)k1c * DIM;
#pragma unroll
        for (int e = 0; e < 16; ++e) {
          int kd = e + (hgrp << 4);          // dh index 0..31 (WMMA K dim)
          bk0[e] = kp0[kd];
          bk1[e] = kp1[kd];
        }
      }
      v8f zero = {};
      v8f s0 = __builtin_amdgcn_wmma_f32_16x16x32_f16(false, aq, false, bk0, (short)0, zero, false, false);
      v8f s1 = __builtin_amdgcn_wmma_f32_16x16x32_f16(false, aq, false, bk1, (short)0, zero, false, false);

      bool ok0 = (kc + lrow) < KK;
      bool ok1 = (kc + 16 + lrow) < KK;
#pragma unroll
      for (int r = 0; r < 8; ++r) {
        s0[r] = ok0 ? s0[r] * SCALE_F : -3.0e38f;
        s1[r] = ok1 ? s1[r] * SCALE_F : -3.0e38f;
      }

      // ---- online softmax: row stats via 16-lane xor-shuffle reductions
#pragma unroll
      for (int r = 0; r < 8; ++r) {
        float mx = fmaxf(s0[r], s1[r]);
#pragma unroll
        for (int off = 8; off >= 1; off >>= 1)
          mx = fmaxf(mx, __shfl_xor(mx, off, 32));
        float nm   = fmaxf(rmax[r], mx);
        float corr = __expf(rmax[r] - nm);
        float p0   = __expf(s0[r] - nm);
        float p1   = __expf(s1[r] - nm);
        float ps = p0 + p1;
#pragma unroll
        for (int off = 8; off >= 1; off >>= 1)
          ps += __shfl_xor(ps, off, 32);
        rsum[r] = rsum[r] * corr + ps;
        rmax[r] = nm;
        o0[r] *= corr;
        o1[r] *= corr;
        s0[r] = p0;
        s1[r] = p1;
      }

      // ---- repack P (C-layout) into WMMA A-layout via LDS
#pragma unroll
      for (int r = 0; r < 8; ++r) {
        int row = r + hgrp*8;
        pbuf[row*32 + lrow]      = (f16_t)s0[r];
        pbuf[row*32 + 16 + lrow] = (f16_t)s1[r];
      }
      __syncthreads();
      v16h ap;
#pragma unroll
      for (int e = 0; e < 16; ++e) {
        int k = e + (hgrp << 4);
        ap[e] = pbuf[lrow*32 + k];
      }
      __syncthreads();

      // ---- wait for the async V DMA, then read WMMA-B operands out of LDS
      asm volatile("s_wait_asynccnt 0" ::: "memory");
      v16h bv0, bv1;
#pragma unroll
      for (int e = 0; e < 16; ++e) {
        int krow = e + (hgrp << 4);        // relative key 0..31
        bv0[e] = vbuf[krow*32 + lrow];
        bv1[e] = vbuf[krow*32 + 16 + lrow];
      }
      // O += P(16x32) @ V(32x32): two 16x16 output tiles over dh
      o0 = __builtin_amdgcn_wmma_f32_16x16x32_f16(false, ap, false, bv0, (short)0, o0, false, false);
      o1 = __builtin_amdgcn_wmma_f32_16x16x32_f16(false, ap, false, bv1, (short)0, o1, false, false);
      // (dscnt wait before the WMMAs guarantees LDS reads complete before the next
      //  chunk's async writes can overwrite vbuf)
    }
  }

  // ---- normalize by softmax denominator and store (b, q, head*32+d) as f16
#pragma unroll
  for (int r = 0; r < 8; ++r) {
    int row = r + hgrp*8;
    int q = qt*16 + row;
    if (q < QQ) {
      float inv = 1.0f / rsum[r];
      f16_t* op = aout + ((size_t)b*QQ + q)*DIM + m*DH;
      op[lrow]      = (f16_t)(o0[r] * inv);
      op[16 + lrow] = (f16_t)(o1[r] * inv);
    }
  }
}

// ---------------- host-side orchestration ----------------
static inline int gemm_blocks(int M, int N) {
  long tiles = (long)((M + 15) / 16) * (N / 64);
  return (int)((tiles + 7) / 8);
}

extern "C" void kernel_launch(void* const* d_in, const int* in_sizes, int n_in,
                              void* d_out, int out_size, void* d_ws, size_t ws_size,
                              hipStream_t stream) {
  const float* x           = (const float*)d_in[0];
  const float* feature     = (const float*)d_in[1];
  const float* I_inv       = (const float*)d_in[2];
  const float* E_inv       = (const float*)d_in[3];
  const float* bev_grid    = (const float*)d_in[4];
  const float* image_plane = (const float*)d_in[5];
  const float* W_cam       = (const float*)d_in[6];
  const float* W_img       = (const float*)d_in[7];
  const float* W_bev       = (const float*)d_in[8];
  const float* b_bev       = (const float*)d_in[9];
  const float* fp_bn_g = (const float*)d_in[10];
  const float* fp_bn_b = (const float*)d_in[11];
  const float* fp_bn_m = (const float*)d_in[12];
  const float* fp_bn_v = (const float*)d_in[13];
  const float* fl_bn_g = (const float*)d_in[14];
  const float* fl_bn_b = (const float*)d_in[15];
  const float* fl_bn_m = (const float*)d_in[16];
  const float* fl_bn_v = (const float*)d_in[17];
  const float* W_fproj = (const float*)d_in[18];
  const float* W_flin  = (const float*)d_in[19];
  const float* q_ln_g  = (const float*)d_in[20];
  const float* q_ln_b  = (const float*)d_in[21];
  const float* Wq      = (const float*)d_in[22];
  const float* bq      = (const float*)d_in[23];
  const float* k_ln_g  = (const float*)d_in[24];
  const float* k_ln_b  = (const float*)d_in[25];
  const float* Wk      = (const float*)d_in[26];
  const float* bk      = (const float*)d_in[27];
  const float* v_ln_g  = (const float*)d_in[28];
  const float* v_ln_b  = (const float*)d_in[29];
  const float* Wv      = (const float*)d_in[30];
  const float* bv      = (const float*)d_in[31];
  const float* Wproj   = (const float*)d_in[32];
  const float* bproj   = (const float*)d_in[33];
  const float* pre_ln_g  = (const float*)d_in[34];
  const float* pre_ln_b  = (const float*)d_in[35];
  const float* mlp_W1    = (const float*)d_in[36];
  const float* mlp_b1    = (const float*)d_in[37];
  const float* mlp_W2    = (const float*)d_in[38];
  const float* mlp_b2    = (const float*)d_in[39];
  const float* post_ln_g = (const float*)d_in[40];
  const float* post_ln_b = (const float*)d_in[41];
  float* out = (float*)d_out;
  (void)in_sizes; (void)n_in; (void)out_size; (void)ws_size;

  // workspace carve-out (deterministic layout; ~80 MB total)
  char* w = (char*)d_ws;
  auto carve = [&](size_t bytes) -> char* {
    char* p = w;
    w += (bytes + 255) & ~(size_t)255;
    return p;
  };
  const size_t NKROWS = (size_t)B_ * NV * KK;   // 20160
  const size_t NQROWS = (size_t)B_ * NV * QQ;   // 30000
  const size_t NZROWS = (size_t)B_ * QQ;        // 5000

  float* c_embed  = (float*)carve((size_t)B_*NV*DIM*4);
  f16_t* qln      = (f16_t*)carve(NQROWS*DIM*2);
  f16_t* frelu_p  = (f16_t*)carve(NKROWS*FD*2);   // reused as valfeat f32 (same byte size)
  f16_t* frelu_l  = (f16_t*)carve(NKROWS*FD*2);
  float* keyfeat  = (float*)carve(NKROWS*DIM*4);
  f16_t* kln      = (f16_t*)carve(NKROWS*DIM*2);
  f16_t* vln      = (f16_t*)carve(NKROWS*DIM*2);
  f16_t* qh       = (f16_t*)carve(NQROWS*DIM*2);
  f16_t* khh      = (f16_t*)carve(NKROWS*DIM*2);
  f16_t* vhh      = (f16_t*)carve(NKROWS*DIM*2);
  f16_t* a_h      = (f16_t*)carve(NZROWS*DIM*2);
  float* z1       = (float*)carve(NZROWS*DIM*4);
  float* z2       = (float*)carve(NZROWS*DIM*4);
  f16_t* z2h      = (f16_t*)carve(NZROWS*DIM*2);
  f16_t* hbuf     = (f16_t*)carve(NZROWS*2*DIM*2);
  float* z3       = (float*)carve(NZROWS*DIM*4);
  f16_t* Wq_h     = (f16_t*)carve(DIM*DIM*2);
  f16_t* Wk_h     = (f16_t*)carve(DIM*DIM*2);
  f16_t* Wv_h     = (f16_t*)carve(DIM*DIM*2);
  f16_t* Wproj_h  = (f16_t*)carve(DIM*DIM*2);
  f16_t* Wfproj_h = (f16_t*)carve(DIM*FD*2);
  f16_t* Wflin_h  = (f16_t*)carve(DIM*FD*2);
  f16_t* W1_h     = (f16_t*)carve(2*DIM*DIM*2);
  f16_t* W2_h     = (f16_t*)carve(DIM*2*DIM*2);
  float* valfeat  = (float*)frelu_p;   // alias: frelu_p dead once keyfeat GEMM ran

  // ---- weights -> f16
  cvt_f16_kernel<<<(DIM*DIM + 255)/256, 256, 0, stream>>>(Wq,     Wq_h,     DIM*DIM);
  cvt_f16_kernel<<<(DIM*DIM + 255)/256, 256, 0, stream>>>(Wk,     Wk_h,     DIM*DIM);
  cvt_f16_kernel<<<(DIM*DIM + 255)/256, 256, 0, stream>>>(Wv,     Wv_h,     DIM*DIM);
  cvt_f16_kernel<<<(DIM*DIM + 255)/256, 256, 0, stream>>>(Wproj,  Wproj_h,  DIM*DIM);
  cvt_f16_kernel<<<(DIM*FD  + 255)/256, 256, 0, stream>>>(W_fproj,Wfproj_h, DIM*FD);
  cvt_f16_kernel<<<(DIM*FD  + 255)/256, 256, 0, stream>>>(W_flin, Wflin_h,  DIM*FD);
  cvt_f16_kernel<<<(2*DIM*DIM+255)/256, 256, 0, stream>>>(mlp_W1, W1_h,     2*DIM*DIM);
  cvt_f16_kernel<<<(2*DIM*DIM+255)/256, 256, 0, stream>>>(mlp_W2, W2_h,     2*DIM*DIM);

  // ---- geometry / BN+ReLU / query build
  cembed_kernel<<<B_*NV, DIM, 0, stream>>>(W_cam, E_inv, c_embed);
  {
    int total = B_*NV*FD*KK;
    frelu_kernel<<<(total + 255)/256, 256, 0, stream>>>(feature,
        fp_bn_g, fp_bn_b, fp_bn_m, fp_bn_v, fl_bn_g, fl_bn_b, fl_bn_m, fl_bn_v,
        frelu_p, frelu_l, total);
  }
  query_kernel<<<(int)NQROWS, DIM, 0, stream>>>(x, bev_grid, W_bev, b_bev, c_embed,
                                                q_ln_g, q_ln_b, qln);

  // ---- 1x1 convs as WMMA GEMMs (Kc = 256)
  gemm_nt_kernel<<<gemm_blocks((int)NKROWS, DIM), 256, 0, stream>>>(
      frelu_p, FD, Wfproj_h, FD, nullptr, keyfeat, nullptr, (int)NKROWS, DIM, FD, 0);
  gemm_nt_kernel<<<gemm_blocks((int)NKROWS, DIM), 256, 0, stream>>>(
      frelu_l, FD, Wflin_h, FD, nullptr, valfeat, nullptr, (int)NKROWS, DIM, FD, 0);

  // ---- q/k/v projections (Kc = 128) -> f16 head-major rows
  gemm_nt_kernel<<<gemm_blocks((int)NQROWS, DIM), 256, 0, stream>>>(
      qln, DIM, Wq_h, DIM, bq, nullptr, qh, (int)NQROWS, DIM, DIM, 0);
  key_ln_kernel<<<(int)NKROWS, DIM, 0, stream>>>(I_inv, E_inv, image_plane, W_img,
                                                 c_embed, keyfeat, k_ln_g, k_ln_b, kln);
  gemm_nt_kernel<<<gemm_blocks((int)NKROWS, DIM), 256, 0, stream>>>(
      kln, DIM, Wk_h, DIM, bk, nullptr, khh, (int)NKROWS, DIM, DIM, 0);
  row_ln_kernel<<<(int)NKROWS, DIM, 0, stream>>>(valfeat, v_ln_g, v_ln_b, vln);
  gemm_nt_kernel<<<gemm_blocks((int)NKROWS, DIM), 256, 0, stream>>>(
      vln, DIM, Wv_h, DIM, bv, nullptr, vhh, (int)NKROWS, DIM, DIM, 0);

  // ---- fused flash attention (softmax over all 6*1680 keys)
  attn_kernel<<<B_*HEADS*QT, 32, 0, stream>>>(qh, khh, vhh, a_h);

  // ---- output projection + skip + pre-LN + MLP + post-LN + transpose
  gemm_nt_kernel<<<gemm_blocks((int)NZROWS, DIM), 256, 0, stream>>>(
      a_h, DIM, Wproj_h, DIM, bproj, z1, nullptr, (int)NZROWS, DIM, DIM, 0);
  addskip_preln_kernel<<<(int)NZROWS, DIM, 0, stream>>>(z1, x, pre_ln_g, pre_ln_b, z2, z2h);
  gemm_nt_kernel<<<gemm_blocks((int)NZROWS, 2*DIM), 256, 0, stream>>>(
      z2h, DIM, W1_h, DIM, mlp_b1, nullptr, hbuf, (int)NZROWS, 2*DIM, DIM, 1 /*gelu*/);
  gemm_nt_kernel<<<gemm_blocks((int)NZROWS, DIM), 256, 0, stream>>>(
      hbuf, 2*DIM, W2_h, 2*DIM, mlp_b2, z3, nullptr, (int)NZROWS, DIM, 2*DIM, 0);
  final_kernel<<<(int)NZROWS, DIM, 0, stream>>>(z2, z3, post_ln_g, post_ln_b, out);
}
